// GCNConvModule_70952859730403
// MI455X (gfx1250) — matrix operand order repaired and checked by reading, not compiled, not run on hardware
//
#include <hip/hip_runtime.h>
#include <hip/hip_bf16.h>

// GCN layer on MI455X (gfx1250, wave32, WMMA).
// out = tanh( D^-1/2 (A|I) D^-1/2 @ (X @ W) + b )
// Both GEMMs run through v_wmma_f32_16x16x32_bf16 with f32 accumulation.
// A-operands are staged in LDS once per block (cooperative convert), then
// each wave pulls its WMMA A-fragment with conflict-free ds_load_b128.

typedef __attribute__((ext_vector_type(16))) __bf16 v16bf;
typedef __attribute__((ext_vector_type(8)))  __bf16 v8bf;
typedef __attribute__((ext_vector_type(4)))  __bf16 v4bf;
typedef __attribute__((ext_vector_type(8)))  float  v8f;
typedef __attribute__((ext_vector_type(4)))  float  v4f;

#define B_    8
#define N_    2048
#define DIN_  512
#define DOUT_ 512
#define KP    512          // K-phase length staged in LDS
#define PITCH 520          // KP + 8 bf16 pad -> 1040B row stride -> 4 banks apart

// float -> bf16 via hardware convert (RNE)
static __device__ __forceinline__ __bf16 f2bf(float x) { return (__bf16)x; }

// ---------------------------------------------------------------------------
// Kernel 1: per-row degree of binary adjacency with forced self-loop,
//           dinv = rsqrt(deg).  One 256-thread block per (b, n) row.
// ---------------------------------------------------------------------------
__global__ void __launch_bounds__(256) gcn_deg_kernel(const float* __restrict__ adj,
                                                      float* __restrict__ dinv) {
    __shared__ int s_cnt;
    const int row = blockIdx.x;                  // b*N + n
    const int n   = row & (N_ - 1);
    const float* arow = adj + (size_t)row * N_;
    if (threadIdx.x == 0) s_cnt = 0;
    __syncthreads();

    const v4f* p = (const v4f*)(arow + threadIdx.x * 8);   // 8 floats / thread
    v4f a0 = p[0], a1 = p[1];
    int local = 0;
#pragma unroll
    for (int i = 0; i < 4; i++) {
        local += (a0[i] != 0.0f);
        local += (a1[i] != 0.0f);
    }
    atomicAdd(&s_cnt, local);
    __syncthreads();

    if (threadIdx.x == 0) {
        const int deg = s_cnt + (arow[n] == 0.0f ? 1 : 0);  // forced self-loop
        dinv[row] = rsqrtf((float)deg);                     // deg >= 1 always
    }
}

// ---------------------------------------------------------------------------
// Kernel 2: W [DIN, DOUT] f32 row-major -> Wt [DOUT, DIN] bf16 row-major
// ---------------------------------------------------------------------------
__global__ void __launch_bounds__(256) gcn_wt_kernel(const float* __restrict__ W,
                                                     __bf16* __restrict__ Wt) {
    const int idx = blockIdx.x * 256 + threadIdx.x;   // over DOUT*DIN
    const int o = idx >> 9;            // / DIN_
    const int f = idx & (DIN_ - 1);
    Wt[idx] = f2bf(W[(size_t)f * DOUT_ + o]);
}

// Assemble a WMMA A-fragment from the LDS strip: lane (lo,hi) takes row lo,
// K range [k0+hi*16, +16) contiguous (matches B operand k order).
static __device__ __forceinline__ v16bf lds_a_frag(const __bf16* sA, int lo, int hi, int k0) {
    const v8bf c0 = *(const v8bf*)(sA + lo * PITCH + k0 + hi * 16);
    const v8bf c1 = *(const v8bf*)(sA + lo * PITCH + k0 + hi * 16 + 8);
    v16bf a;
#pragma unroll
    for (int i = 0; i < 8; i++) { a[i] = c0[i]; a[8 + i] = c1[i]; }
    return a;
}

// ---------------------------------------------------------------------------
// Kernel 3: h = X @ W, stored TRANSPOSED as bf16: hT[b][o][m].
// Block = 8 waves over one 16-row m-tile; wave w owns 64 output columns.
// X strip (16 x 512) converted to bf16 in LDS cooperatively, once.
// ---------------------------------------------------------------------------
__global__ void __launch_bounds__(256) gcn_gemm1_kernel(const float* __restrict__ X,
                                                        const __bf16* __restrict__ Wt,
                                                        __bf16* __restrict__ hT) {
    __shared__ __bf16 sX[16 * PITCH];
    const int b    = blockIdx.x >> 7;         // / 128 m-tiles
    const int mt   = blockIdx.x & 127;
    const int t    = threadIdx.x;
    const int wave = t >> 5;
    const int lane = t & 31;
    const int lo   = lane & 15;
    const int hi   = lane >> 4;
    const int m0   = mt * 16;
    const int o_base = wave * 64;

    // Cooperative fill: 16 rows x 512 cols = 2048 float4, 8 per thread, coalesced
    const float* xbase = X + (size_t)(b * N_ + m0) * DIN_;
#pragma unroll
    for (int j = 0; j < 8; j++) {
        const int f4  = t + 256 * j;
        const int row = f4 >> 7;              // 128 float4 per row
        const int c   = (f4 & 127) * 4;
        v4f x = *(const v4f*)(xbase + row * DIN_ + c);
        v4bf h;
#pragma unroll
        for (int i = 0; i < 4; i++) h[i] = f2bf(x[i]);
        *(v4bf*)(sX + row * PITCH + c) = h;
    }
    __syncthreads();

    v8f acc[4] = {};
    for (int k0 = 0; k0 < DIN_; k0 += 32) {
        const v16bf a = lds_a_frag(sX, lo, hi, k0);
#pragma unroll
        for (int nt = 0; nt < 4; nt++) {
            const int o = o_base + nt * 16 + lo;
            // B tile 32x16: lane = column o, k-range [k0 + hi*16, +16) contiguous
            v16bf bv = *(const v16bf*)(Wt + (size_t)o * DIN_ + k0 + hi * 16);
            acc[nt] = __builtin_amdgcn_wmma_f32_16x16x32_bf16(
                false, a, false, bv, (short)0, acc[nt], false, false);
        }
    }

    // D layout: VGPR r holds (m = m0 + r + 8*hi, n = o). Transposed store:
    // 8 consecutive m per lane -> one packed 16-byte bf16 store.
#pragma unroll
    for (int nt = 0; nt < 4; nt++) {
        const int o = o_base + nt * 16 + lo;
        v8bf h;
#pragma unroll
        for (int r = 0; r < 8; r++) h[r] = f2bf(acc[nt][r]);
        *(v8bf*)(hT + (size_t)(b * DOUT_ + o) * N_ + m0 + hi * 8) = h;
    }
}

// ---------------------------------------------------------------------------
// Kernel 4: agg = norm @ h ; out = tanh(agg + bias).
// norm strip (16 x 2048) built in LDS in 4 K-phases of 512:
//   norm[i][k] = dinv_i * dinv_k * (adj!=0 || k==i)
// ---------------------------------------------------------------------------
__global__ void __launch_bounds__(256) gcn_gemm2_kernel(const float* __restrict__ adj,
                                                        const float* __restrict__ dinv,
                                                        const __bf16* __restrict__ hT,
                                                        const float* __restrict__ bias,
                                                        float* __restrict__ out) {
    __shared__ __bf16 sA[16 * PITCH];
    const int b    = blockIdx.x >> 7;
    const int mt   = blockIdx.x & 127;
    const int t    = threadIdx.x;
    const int wave = t >> 5;
    const int lane = t & 31;
    const int lo   = lane & 15;
    const int hi   = lane >> 4;
    const int m0   = mt * 16;
    const int o_base = wave * 64;

    const float* adjbase = adj + (size_t)(b * N_ + m0) * N_;
    const float* dv      = dinv + b * N_;

    v8f acc[4] = {};
    for (int p = 0; p < N_ / KP; p++) {
        const int kbase = p * KP;
        // Cooperative fill of normalized-adjacency slab (16 x 512 bf16)
#pragma unroll
        for (int j = 0; j < 8; j++) {
            const int f4  = t + 256 * j;
            const int row = f4 >> 7;
            const int c   = (f4 & 127) * 4;
            const int kg0 = kbase + c;
            v4f av = *(const v4f*)(adjbase + (size_t)row * N_ + kg0);
            v4f dk = *(const v4f*)(dv + kg0);
            const float di = dv[m0 + row];
            v4bf nv;
#pragma unroll
            for (int i = 0; i < 4; i++) {
                const float val = (av[i] != 0.0f || (kg0 + i) == (m0 + row)) ? di * dk[i] : 0.0f;
                nv[i] = f2bf(val);
            }
            *(v4bf*)(sA + row * PITCH + c) = nv;
        }
        __syncthreads();

        for (int kc = 0; kc < KP; kc += 32) {
            const v16bf a = lds_a_frag(sA, lo, hi, kc);
#pragma unroll
            for (int nt = 0; nt < 4; nt++) {
                const int o = o_base + nt * 16 + lo;
                v16bf bv = *(const v16bf*)(hT + (size_t)(b * DOUT_ + o) * N_ + kbase + kc + hi * 16);
                acc[nt] = __builtin_amdgcn_wmma_f32_16x16x32_bf16(
                    false, a, false, bv, (short)0, acc[nt], false, false);
            }
        }
        __syncthreads();   // protect slab before next phase's fill
    }

#pragma unroll
    for (int nt = 0; nt < 4; nt++) {
        const int o = o_base + nt * 16 + lo;
        const float bo = bias[o];
#pragma unroll
        for (int r = 0; r < 8; r++) {
            const int m = m0 + hi * 8 + r;
            out[(size_t)(b * N_ + m) * DOUT_ + o] = tanhf(acc[nt][r] + bo);
        }
    }
}

// ---------------------------------------------------------------------------
extern "C" void kernel_launch(void* const* d_in, const int* in_sizes, int n_in,
                              void* d_out, int out_size, void* d_ws, size_t ws_size,
                              hipStream_t stream) {
    const float* X    = (const float*)d_in[0];   // [B, N, DIN]
    const float* adj  = (const float*)d_in[1];   // [B, N, N]
    const float* W    = (const float*)d_in[2];   // [DIN, DOUT]
    const float* bias = (const float*)d_in[3];   // [DOUT]
    float* out = (float*)d_out;                  // [B, N, DOUT]

    char* ws = (char*)d_ws;
    float*  dinv = (float*)ws;                                 // 8*2048*4   = 64 KB
    __bf16* Wt   = (__bf16*)(ws + (64u << 10));                // 512*512*2  = 512 KB
    __bf16* hT   = (__bf16*)(ws + (64u << 10) + (512u << 10)); // 8*512*2048*2 = 16 MB

    gcn_deg_kernel  <<<B_ * N_,              256, 0, stream>>>(adj, dinv);
    gcn_wt_kernel   <<<(DOUT_ * DIN_) / 256, 256, 0, stream>>>(W, Wt);
    gcn_gemm1_kernel<<<B_ * (N_ / 16),       256, 0, stream>>>(X, Wt, hT);
    gcn_gemm2_kernel<<<B_ * (N_ / 16),       256, 0, stream>>>(adj, dinv, hT, bias, out);
}
